// LIDM_5428838662204
// MI455X (gfx1250) — matrix operand order repaired and checked by reading, not compiled, not run on hardware
//
#include <hip/hip_runtime.h>
#include <hip/hip_bf16.h>

// ---------------------------------------------------------------------------
// LIDM on MI455X (gfx1250, wave32, WMMA, TDM).
//
// obsr_enable==1 fast path: the g-MLP is dead code and the scan's only live
// recurrence is the GRU hidden state, so:
//   Phase 1 (parallel WMMA GEMMs): ZX[t] = cz*fx(obsrv[1+t]) + a*sx*eps_z[t]
//   Phase 2 (persistent 1-WGP kernel): bidirectional 2-layer GRU + fc
//     - zx[t] staged in LDS, double-buffered, prefetched one step ahead by
//       the Tensor Data Mover (tensor_load_to_lds + s_wait_tensorcnt)
//     - layer-0 concat (inp0) kept in LDS; h state double-buffered in global
//       (L2-resident); out[1+t] written directly.
//   d_out[0] = sx*noise_z0.
// All matrix math: v_wmma_f32_16x16x32_f16 (f16 A/B, f32 accumulate).
// ---------------------------------------------------------------------------

typedef _Float16 h16;
typedef __attribute__((ext_vector_type(16))) _Float16 v16h;
typedef __attribute__((ext_vector_type(8)))  _Float16 v8h;
typedef __attribute__((ext_vector_type(8)))  float    v8f;
typedef __attribute__((ext_vector_type(4)))  unsigned int ui4;
typedef __attribute__((ext_vector_type(8)))  int      i8v;
typedef __attribute__((ext_vector_type(4)))  int      i4v;

static constexpr int T  = 1024;
static constexpr int TS = 1023;   // steps in the scan
static constexpr int Bz = 256;    // batch
static constexpr int L  = 128;    // latent
static constexpr int O  = 256;    // obs
static constexpr int CH = 256;    // phase-1 timestep chunk

static constexpr float SX    = 0.1f;
static constexpr float ASX   = 0.05f;            // alpha*sx
static constexpr float CZ    = 0.99874921777f;   // sqrt(1-(alpha*sx)^2)
static constexpr float LRELU = 0.01f;

#define DEVINL __device__ __forceinline__

DEVINL int lane_id() { return threadIdx.x & 31; }

DEVINL v8f wmma16(v16h a, v16h b, v8f c) {
  return __builtin_amdgcn_wmma_f32_16x16x32_f16(
      /*neg_a=*/false, a, /*neg_b=*/false, b,
      /*c_mod=*/(short)0, c, /*reuse_a=*/false, /*reuse_b=*/false);
}

DEVINL v8f vzero8() {
  v8f c = {0.f,0.f,0.f,0.f,0.f,0.f,0.f,0.f};
  return c;
}

// A fragment (16x32, f16, row-major X with leading dim ld).
// ISA layout: lanes 0-15 (M=lane): halves0-7 = K k0..k0+7, halves8-15 = k0+16..23
//             lanes16-31 (M=lane-16): k0+8..15 and k0+24..31.
DEVINL v16h load_a16(const h16* X, int ld, int m0, int k0) {
  const int lane = lane_id();
  const h16* p = X + (size_t)(m0 + (lane & 15)) * ld + k0 + ((lane >> 4) << 3);
  v8h lo = *(const v8h*)p;
  v8h hi = *(const v8h*)(p + 16);
  v16h a;
#pragma unroll
  for (int i = 0; i < 8; ++i) { a[i] = lo[i]; a[8 + i] = hi[i]; }
  return a;
}

// Same A fragment, but source is f32 (first fx layer reads obsrv directly).
DEVINL v16h load_a32(const float* X, int ld, int m0, int k0) {
  const int lane = lane_id();
  const float* p = X + (size_t)(m0 + (lane & 15)) * ld + k0 + ((lane >> 4) << 3);
  v8f lo = *(const v8f*)p;
  v8f hi = *(const v8f*)(p + 16);
  v16h a;
#pragma unroll
  for (int i = 0; i < 8; ++i) { a[i] = (h16)lo[i]; a[8 + i] = (h16)hi[i]; }
  return a;
}

// B fragment (32x16). B[k][n] = W[n][k], W row-major [N, ldk].
// lanes 0-15: n=lane, K k0..k0+7, k0+16..23 ; lanes16-31: n=lane-16, K k0+8..15, k0+24..31.
DEVINL v16h load_b16(const h16* W, int ldk, int n0, int k0) {
  const int lane = lane_id();
  const h16* p = W + (size_t)(n0 + (lane & 15)) * ldk + k0 + ((lane >> 4) << 3);
  v8h lo = *(const v8h*)p;
  v8h hi = *(const v8h*)(p + 16);
  v16h b;
#pragma unroll
  for (int i = 0; i < 8; ++i) { b[i] = lo[i]; b[8 + i] = hi[i]; }
  return b;
}

DEVINL float sigm(float x) { return 1.0f / (1.0f + __expf(-x)); }

// TDM: DMA one zx step tile ([256 rows x 128 halves], row-major, contiguous)
// from global memory into LDS. D# per CDNA5 ISA:
//   group0: count=1 | lds_addr[63:32] | global_addr[120:64] | type=2
//   group1: data_size=2B, tensor_dim0=128, tensor_dim1=256, tile 128x256,
//           tensor_dim0_stride=128, no multicast / padding / iterate.
// This toolchain exposes the 6-arg builtin (g0, g1, g2, g3, g4, cpol).
DEVINL void tdm_load_zx(const h16* gsrc, unsigned lds_off) {
  unsigned long long ga = (unsigned long long)(uintptr_t)gsrc;
  ui4 g0;
  g0[0] = 1u;                                               // count=1 (user D#)
  g0[1] = lds_off;                                          // LDS byte address
  g0[2] = (unsigned)(ga & 0xFFFFFFFFu);                     // global addr lo
  g0[3] = (unsigned)((ga >> 32) & 0x01FFFFFFu) | (2u << 30);// addr hi | type=2
  i8v g1;
  g1[0] = (int)(1u << 16);     // workgroup_mask=0, data_size=1 (2 bytes)
  g1[1] = (int)(128u << 16);   // tensor_dim0[15:0] = 128 at bits [63:48]
  g1[2] = (int)(256u << 16);   // tensor_dim0[31:16]=0, tensor_dim1[15:0]=256
  g1[3] = (int)(128u << 16);   // tensor_dim1[31:16]=0, tile_dim0=128
  g1[4] = 256;                 // tile_dim1=256, tile_dim2=0
  g1[5] = 128;                 // tensor_dim0_stride lo32 = 128
  g1[6] = 0;                   // stride hi / tensor_dim1_stride lo
  g1[7] = 0;
  i4v gz4 = {0, 0, 0, 0};
  i8v gz8 = {0, 0, 0, 0, 0, 0, 0, 0};
  __builtin_amdgcn_tensor_load_to_lds(g0, g1, gz4, gz4, gz8, 0);
}

// ---------------------------------------------------------------------------
// Elementwise helpers
// ---------------------------------------------------------------------------
__global__ void k_f32_to_f16(const float* __restrict__ src, h16* __restrict__ dst, int n) {
  int i = blockIdx.x * blockDim.x + threadIdx.x;
  if (i < n) dst[i] = (h16)src[i];
}

__global__ void k_out0(const float* __restrict__ nz, float* __restrict__ out) {
  int i = blockIdx.x * blockDim.x + threadIdx.x;  // n = B*L
  out[i] = SX * nz[i];
}

__global__ void k_hinit(const float* __restrict__ nh,
                        h16* h00, h16* h01, h16* h10, h16* h11) {
  int i = blockIdx.x * blockDim.x + threadIdx.x;  // n = 4*B*L
  int which = i / (Bz * L);
  int j = i - which * (Bz * L);
  h16 v = (h16)(SX * nh[i]);
  h16* dst = (which == 0) ? h00 : (which == 1) ? h01 : (which == 2) ? h10 : h11;
  dst[j] = v;
}

// ---------------------------------------------------------------------------
// Phase-1 GEMM kernels. One 16x16 output tile per wave, 8 waves per block.
// Y[R,N] = act(X[R,K] @ W[N,K]^T + bias)
// ---------------------------------------------------------------------------
__global__ void k_gemm_lrelu_f32a(const float* __restrict__ X, const h16* __restrict__ W,
                                  const float* __restrict__ bias, h16* __restrict__ Y,
                                  int R, int K, int N) {
  const int ntiles = N >> 4;
  const int tile = blockIdx.x * 8 + (threadIdx.x >> 5);
  if (tile >= (R >> 4) * ntiles) return;
  const int mt = tile / ntiles, nt = tile % ntiles;
  v8f c = vzero8();
  for (int k = 0; k < K; k += 32)
    c = wmma16(load_a32(X, K, mt * 16, k), load_b16(W, K, nt * 16, k), c);
  const int lane = lane_id();
  const int col = nt * 16 + (lane & 15);
  const int row0 = mt * 16 + ((lane >> 4) << 3);
  const float bv = bias[col];
#pragma unroll
  for (int r = 0; r < 8; ++r) {
    float y = c[r] + bv;
    y = y > 0.f ? y : LRELU * y;
    Y[(size_t)(row0 + r) * N + col] = (h16)y;
  }
}

__global__ void k_gemm_lrelu_f16a(const h16* __restrict__ X, const h16* __restrict__ W,
                                  const float* __restrict__ bias, h16* __restrict__ Y,
                                  int R, int K, int N) {
  const int ntiles = N >> 4;
  const int tile = blockIdx.x * 8 + (threadIdx.x >> 5);
  if (tile >= (R >> 4) * ntiles) return;
  const int mt = tile / ntiles, nt = tile % ntiles;
  v8f c = vzero8();
  for (int k = 0; k < K; k += 32)
    c = wmma16(load_a16(X, K, mt * 16, k), load_b16(W, K, nt * 16, k), c);
  const int lane = lane_id();
  const int col = nt * 16 + (lane & 15);
  const int row0 = mt * 16 + ((lane >> 4) << 3);
  const float bv = bias[col];
#pragma unroll
  for (int r = 0; r < 8; ++r) {
    float y = c[r] + bv;
    y = y > 0.f ? y : LRELU * y;
    Y[(size_t)(row0 + r) * N + col] = (h16)y;
  }
}

// Final fx layer: ZX = cz*(X @ W4^T + b4) + a*sx*eps,  N = L = 128.
__global__ void k_fx_out(const h16* __restrict__ X, const h16* __restrict__ W,
                         const float* __restrict__ bias, const float* __restrict__ eps,
                         h16* __restrict__ ZX, int R) {
  const int ntiles = L >> 4;  // 8
  const int tile = blockIdx.x * 8 + (threadIdx.x >> 5);
  if (tile >= (R >> 4) * ntiles) return;
  const int mt = tile / ntiles, nt = tile % ntiles;
  v8f c = vzero8();
  for (int k = 0; k < O; k += 32)
    c = wmma16(load_a16(X, O, mt * 16, k), load_b16(W, O, nt * 16, k), c);
  const int lane = lane_id();
  const int col = nt * 16 + (lane & 15);
  const int row0 = mt * 16 + ((lane >> 4) << 3);
  const float bv = bias[col];
#pragma unroll
  for (int r = 0; r < 8; ++r) {
    size_t idx = (size_t)(row0 + r) * L + col;
    ZX[idx] = (h16)(CZ * (c[r] + bv) + ASX * eps[idx]);
  }
}

// ---------------------------------------------------------------------------
// Phase-2: persistent single-workgroup GRU scan (512 threads = 16 waves).
// LDS: zx double buffer (2 x 64 KB, TDM-prefetched) + inp0 (128 KB) = 256 KB.
// GRU weights (~1 MB f16) stay hot in L2; h state double-buffered in global.
// ---------------------------------------------------------------------------
struct GruArgs {
  const h16* zxall;                      // [TS,B,L] f16
  const h16* wih0[2]; const h16* whh0[2];
  const float* bih0[2]; const float* bhh0[2];
  const h16* wih1[2]; const h16* whh1[2];
  const float* bih1[2]; const float* bhh1[2];
  const h16* fcw; const float* fcb;
  h16* h0[2][2];                          // [buf][dir], [B,L] f16
  h16* h1[2][2];
  h16* inp1;                              // [B,2L] f16 (layer1 concat, global)
  float* out;                             // d_out base, [T,B,L] f32
};

static constexpr int NW = 16;  // waves in the persistent workgroup
static constexpr unsigned LDS_ZX0  = 0;                    // zx buf 0
static constexpr unsigned LDS_ZX1  = Bz * L * 2;           // zx buf 1
static constexpr unsigned LDS_INP0 = 2 * Bz * L * 2;       // inp0
static constexpr unsigned LDS_BYTES = LDS_INP0 + Bz * 2 * L * 2;  // 256 KB

__global__ __launch_bounds__(NW * 32)
void k_gru_serial(GruArgs ga) {
  extern __shared__ h16 smem[];
  const int wave = threadIdx.x >> 5;
  const int lane = lane_id();
  // NOTE: no pointer *arrays* of LDS-derived pointers here — an aggregate
  // initializer of addrspace(3) casts becomes an unlinkable static init.
  h16* const zxbuf0 = smem;
  h16* const zxbuf1 = smem + Bz * L;
  h16* const inp0   = smem + 2 * Bz * L;
  const unsigned lds_base = (unsigned)(uintptr_t)smem;  // LDS aperture: low 32b = offset
  int cur = 0;

  // Prime the pipeline: TDM-load zx[0] into buffer 0.
  if (wave == 0) tdm_load_zx(ga.zxall, lds_base + LDS_ZX0);

  for (int t = 0; t < TS; ++t) {
    // Prefetch next step's zx into the other LDS buffer, then retire the
    // load for THIS step (<=1 outstanding keeps the prefetch in flight).
    if (wave == 0) {
      if (t + 1 < TS) {
        tdm_load_zx(ga.zxall + (size_t)(t + 1) * Bz * L,
                    lds_base + (((t + 1) & 1) ? LDS_ZX1 : LDS_ZX0));
        __builtin_amdgcn_s_wait_tensorcnt(1);
      } else {
        __builtin_amdgcn_s_wait_tensorcnt(0);
      }
    }
    __syncthreads();
    const h16* zx = (t & 1) ? zxbuf1 : zxbuf0;

    // ---- Stage A: GRU layer 0, both directions. 2*16*8 = 256 tiles.
    for (int tile = wave; tile < 256; tile += NW) {
      const int dir = tile >> 7;
      const int mt  = (tile >> 3) & 15;
      const int nt  = tile & 7;
      const h16* Wih = ga.wih0[dir];
      const h16* Whh = ga.whh0[dir];
      const h16* hp  = ga.h0[cur][dir];
      v8f ir = vzero8(), iz = vzero8(), in_ = vzero8();
      v8f hr = vzero8(), hz = vzero8(), hn = vzero8();
      for (int k = 0; k < L; k += 32) {
        v16h az = load_a16(zx, L, mt * 16, k);   // LDS (ds_load_b128)
        v16h ah = load_a16(hp, L, mt * 16, k);   // global, L2-hot
        ir  = wmma16(az, load_b16(Wih, L, nt * 16,          k), ir);
        iz  = wmma16(az, load_b16(Wih, L, nt * 16 + L,      k), iz);
        in_ = wmma16(az, load_b16(Wih, L, nt * 16 + 2 * L,  k), in_);
        hr  = wmma16(ah, load_b16(Whh, L, nt * 16,          k), hr);
        hz  = wmma16(ah, load_b16(Whh, L, nt * 16 + L,      k), hz);
        hn  = wmma16(ah, load_b16(Whh, L, nt * 16 + 2 * L,  k), hn);
      }
      const int col = nt * 16 + (lane & 15);
      const int row0 = mt * 16 + ((lane >> 4) << 3);
      const float bir = ga.bih0[dir][col],     bhr = ga.bhh0[dir][col];
      const float biz = ga.bih0[dir][L + col], bhz = ga.bhh0[dir][L + col];
      const float bin = ga.bih0[dir][2*L+col], bhn = ga.bhh0[dir][2*L+col];
      h16* hnx = ga.h0[cur ^ 1][dir];
#pragma unroll
      for (int r = 0; r < 8; ++r) {
        float rg = sigm(ir[r] + bir + hr[r] + bhr);
        float zg = sigm(iz[r] + biz + hz[r] + bhz);
        float ng = tanhf(in_[r] + bin + rg * (hn[r] + bhn));
        float ho = (float)hp[(size_t)(row0 + r) * L + col];
        h16 hv = (h16)((1.0f - zg) * ng + zg * ho);
        hnx[(size_t)(row0 + r) * L + col] = hv;
        inp0[(size_t)(row0 + r) * (2 * L) + dir * L + col] = hv;  // LDS store
      }
    }
    __threadfence_block();
    __syncthreads();

    // ---- Stage B: GRU layer 1 (input = concat[hf,hb] in LDS, K=256 for gi).
    for (int tile = wave; tile < 256; tile += NW) {
      const int dir = tile >> 7;
      const int mt  = (tile >> 3) & 15;
      const int nt  = tile & 7;
      const h16* Wih = ga.wih1[dir];
      const h16* Whh = ga.whh1[dir];
      const h16* hp  = ga.h1[cur][dir];
      v8f ir = vzero8(), iz = vzero8(), in_ = vzero8();
      v8f hr = vzero8(), hz = vzero8(), hn = vzero8();
      for (int k = 0; k < 2 * L; k += 32) {
        v16h ax = load_a16(inp0, 2 * L, mt * 16, k);   // LDS
        ir  = wmma16(ax, load_b16(Wih, 2 * L, nt * 16,         k), ir);
        iz  = wmma16(ax, load_b16(Wih, 2 * L, nt * 16 + L,     k), iz);
        in_ = wmma16(ax, load_b16(Wih, 2 * L, nt * 16 + 2 * L, k), in_);
      }
      for (int k = 0; k < L; k += 32) {
        v16h ah = load_a16(hp, L, mt * 16, k);
        hr = wmma16(ah, load_b16(Whh, L, nt * 16,         k), hr);
        hz = wmma16(ah, load_b16(Whh, L, nt * 16 + L,     k), hz);
        hn = wmma16(ah, load_b16(Whh, L, nt * 16 + 2 * L, k), hn);
      }
      const int col = nt * 16 + (lane & 15);
      const int row0 = mt * 16 + ((lane >> 4) << 3);
      const float bir = ga.bih1[dir][col],     bhr = ga.bhh1[dir][col];
      const float biz = ga.bih1[dir][L + col], bhz = ga.bhh1[dir][L + col];
      const float bin = ga.bih1[dir][2*L+col], bhn = ga.bhh1[dir][2*L+col];
      h16* hnx = ga.h1[cur ^ 1][dir];
#pragma unroll
      for (int r = 0; r < 8; ++r) {
        float rg = sigm(ir[r] + bir + hr[r] + bhr);
        float zg = sigm(iz[r] + biz + hz[r] + bhz);
        float ng = tanhf(in_[r] + bin + rg * (hn[r] + bhn));
        float ho = (float)hp[(size_t)(row0 + r) * L + col];
        h16 hv = (h16)((1.0f - zg) * ng + zg * ho);
        hnx[(size_t)(row0 + r) * L + col] = hv;
        ga.inp1[(size_t)(row0 + r) * (2 * L) + dir * L + col] = hv;
      }
    }
    __threadfence_block();
    __syncthreads();

    // ---- Stage C: out[1+t] = inp1 @ fcW^T + fcb   (128 tiles, f32 out)
    float* outp = ga.out + (size_t)(t + 1) * Bz * L;
    for (int tile = wave; tile < 128; tile += NW) {
      const int mt = tile >> 3, nt = tile & 7;
      v8f c = vzero8();
      for (int k = 0; k < 2 * L; k += 32)
        c = wmma16(load_a16(ga.inp1, 2 * L, mt * 16, k),
                   load_b16(ga.fcw, 2 * L, nt * 16, k), c);
      const int col = nt * 16 + (lane & 15);
      const int row0 = mt * 16 + ((lane >> 4) << 3);
      const float bv = ga.fcb[col];
#pragma unroll
      for (int r = 0; r < 8; ++r)
        outp[(size_t)(row0 + r) * L + col] = c[r] + bv;
    }
    __threadfence_block();
    __syncthreads();
    cur ^= 1;
  }
}

// ---------------------------------------------------------------------------
// Host side
// ---------------------------------------------------------------------------
extern "C" void kernel_launch(void* const* d_in, const int* in_sizes, int n_in,
                              void* d_out, int out_size, void* d_ws, size_t ws_size,
                              hipStream_t stream) {
  (void)in_sizes; (void)n_in; (void)out_size; (void)ws_size;
  int i = 0;
  const float* obsrv = (const float*)d_in[i++];                    // [1024,256,256]
  const float *fxW[5], *fxB[5];
  for (int l = 0; l < 5; ++l) { fxW[l] = (const float*)d_in[i++]; fxB[l] = (const float*)d_in[i++]; }
  for (int l = 0; l < 10; ++l) { (void)d_in[i++]; }                // g params: dead (obsr_enable=1)
  const float *gWih[2][2], *gWhh[2][2], *gBih[2][2], *gBhh[2][2];
  for (int l = 0; l < 2; ++l)
    for (int d = 0; d < 2; ++d) {
      gWih[l][d] = (const float*)d_in[i++];
      gWhh[l][d] = (const float*)d_in[i++];
      gBih[l][d] = (const float*)d_in[i++];
      gBhh[l][d] = (const float*)d_in[i++];
    }
  const float* fcW = (const float*)d_in[i++];
  const float* fcb = (const float*)d_in[i++];
  const float* noise_z0 = (const float*)d_in[i++];
  const float* noise_h0 = (const float*)d_in[i++];
  const float* eps_z = (const float*)d_in[i++];
  const float* eps_x = (const float*)d_in[i++]; (void)eps_x;       // dead (obsr_enable=1)
  (void)d_in[i++];                                                  // obsr_enable (static 1)

  float* out = (float*)d_out;

  // Workspace carve-out (~134 MB total).
  char* wsb = (char*)d_ws; size_t off = 0;
  auto carve = [&](size_t bytes) -> void* {
    void* p = wsb + off; off = (off + bytes + 255) & ~(size_t)255; return p;
  };
  h16* wZX  = (h16*)carve((size_t)TS * Bz * L * 2);
  h16* wB0  = (h16*)carve((size_t)CH * Bz * O * 2);
  h16* wB1  = (h16*)carve((size_t)CH * Bz * O * 2);
  h16* wFx[5];
  const int fxN[5] = {O*O, O*O, O*O, O*O, L*O};
  for (int l = 0; l < 5; ++l) wFx[l] = (h16*)carve((size_t)fxN[l] * 2);
  h16 *wIh0[2], *wHh0[2], *wIh1[2], *wHh1[2];
  for (int d = 0; d < 2; ++d) { wIh0[d] = (h16*)carve(3*L*L*2);   wHh0[d] = (h16*)carve(3*L*L*2); }
  for (int d = 0; d < 2; ++d) { wIh1[d] = (h16*)carve(3*L*2*L*2); wHh1[d] = (h16*)carve(3*L*L*2); }
  h16* wFc = (h16*)carve((size_t)L * 2 * L * 2);
  h16 *h0b[2][2], *h1b[2][2];
  for (int b = 0; b < 2; ++b)
    for (int d = 0; d < 2; ++d) { h0b[b][d] = (h16*)carve(Bz*L*2); h1b[b][d] = (h16*)carve(Bz*L*2); }
  h16* inp1 = (h16*)carve(Bz * 2 * L * 2);

  // Weight f32 -> f16 conversions.
  auto cvt = [&](const float* s, h16* d, int n) {
    k_f32_to_f16<<<(n + 255) / 256, 256, 0, stream>>>(s, d, n);
  };
  for (int l = 0; l < 5; ++l) cvt(fxW[l], wFx[l], fxN[l]);
  for (int d = 0; d < 2; ++d) {
    cvt(gWih[0][d], wIh0[d], 3*L*L);   cvt(gWhh[0][d], wHh0[d], 3*L*L);
    cvt(gWih[1][d], wIh1[d], 3*L*2*L); cvt(gWhh[1][d], wHh1[d], 3*L*L);
  }
  cvt(fcW, wFc, L * 2 * L);

  // Output row 0 and initial hidden state.
  k_out0<<<(Bz * L) / 256, 256, 0, stream>>>(noise_z0, out);
  k_hinit<<<(4 * Bz * L) / 256, 256, 0, stream>>>(noise_h0,
      h0b[0][0], h0b[0][1], h1b[0][0], h1b[0][1]);

  // Phase 1: fx MLP over all timesteps (chunked), producing ZX.
  auto blocks = [](int rows, int n) { return ((rows >> 4) * (n >> 4) + 7) / 8; };
  for (int c = 0; c < 4; ++c) {
    const int t0 = c * CH;
    const int cs = (t0 + CH <= TS) ? CH : (TS - t0);
    const int rows = cs * Bz;
    const float* xin = obsrv + (size_t)(1 + t0) * Bz * O;
    k_gemm_lrelu_f32a<<<blocks(rows, O), 256, 0, stream>>>(xin,  wFx[0], fxB[0], wB0, rows, O, O);
    k_gemm_lrelu_f16a<<<blocks(rows, O), 256, 0, stream>>>(wB0, wFx[1], fxB[1], wB1, rows, O, O);
    k_gemm_lrelu_f16a<<<blocks(rows, O), 256, 0, stream>>>(wB1, wFx[2], fxB[2], wB0, rows, O, O);
    k_gemm_lrelu_f16a<<<blocks(rows, O), 256, 0, stream>>>(wB0, wFx[3], fxB[3], wB1, rows, O, O);
    k_fx_out<<<blocks(rows, L), 256, 0, stream>>>(wB1, wFx[4], fxB[4],
        eps_z + (size_t)t0 * Bz * L, wZX + (size_t)t0 * Bz * L, rows);
  }

  // Phase 2: persistent serial GRU scan with TDM zx prefetch (256 KB LDS).
  (void)hipFuncSetAttribute(reinterpret_cast<const void*>(k_gru_serial),
                            hipFuncAttributeMaxDynamicSharedMemorySize, (int)LDS_BYTES);
  GruArgs ga;
  ga.zxall = wZX;
  for (int d = 0; d < 2; ++d) {
    ga.wih0[d] = wIh0[d]; ga.whh0[d] = wHh0[d];
    ga.bih0[d] = gBih[0][d]; ga.bhh0[d] = gBhh[0][d];
    ga.wih1[d] = wIh1[d]; ga.whh1[d] = wHh1[d];
    ga.bih1[d] = gBih[1][d]; ga.bhh1[d] = gBhh[1][d];
  }
  ga.fcw = wFc; ga.fcb = fcb;
  for (int b = 0; b < 2; ++b)
    for (int d = 0; d < 2; ++d) { ga.h0[b][d] = h0b[b][d]; ga.h1[b][d] = h1b[b][d]; }
  ga.inp1 = inp1; ga.out = out;
  k_gru_serial<<<1, NW * 32, LDS_BYTES, stream>>>(ga);
}
